// LocalAttention_14929306321699
// MI455X (gfx1250) — compile-verified, matrix operand order checked by reading, not run on hardware
//
#include <hip/hip_runtime.h>
#include <hip/hip_bf16.h>

typedef __attribute__((ext_vector_type(16))) _Float16     v16h;
typedef __attribute__((ext_vector_type(8)))  float        v8f;
typedef __attribute__((ext_vector_type(4)))  unsigned int v4u;
typedef __attribute__((ext_vector_type(2)))  unsigned int v2u;
typedef __attribute__((ext_vector_type(4)))  int          v4i;
typedef __attribute__((address_space(1))) v4i* gv4i_p;  // global ("__device__")
typedef __attribute__((address_space(3))) v4i* lv4i_p;  // LDS ("__shared__")

#define S_LEN  4096
#define BATCH  8
#define DMODEL 768
#define NHEAD  12
#define HD     64
#define WIN    128
#define NWIN   32
#define W3     384
#define NEGMAX (-3.402823466e+38f)

#if __has_builtin(__builtin_amdgcn_global_load_async_to_lds_b128)
#define HAVE_ASYNC_LDS 1
#endif

union Frag { v4u u[2]; v16h h; };

__device__ __forceinline__ v8f wmma_f16(v16h a, v16h b, v8f c) {
  return __builtin_amdgcn_wmma_f32_16x16x32_f16(false, a, false, b, (short)0, c,
                                                false, false);
}

__device__ __forceinline__ void wait_async0() {
#ifdef HAVE_ASYNC_LDS
#if __has_builtin(__builtin_amdgcn_s_wait_asynccnt)
  __builtin_amdgcn_s_wait_asynccnt(0);
#else
  asm volatile("s_wait_asynccnt 0" ::: "memory");
#endif
#endif
}

// ---------------------------------------------------------------------------
// Kernel 1: fused Q/K/V projection GEMM (z selects weight), f32 in -> f16 out
// scattered into (B*H, S, hd) layout. Block tile 128x128, K-chunks of 32,
// double-buffered LDS so global loads overlap the WMMAs.
// ---------------------------------------------------------------------------
__global__ __launch_bounds__(256) void qkv_gemm_wmma(
    const float* __restrict__ X, const float* __restrict__ W0,
    const float* __restrict__ W1, const float* __restrict__ W2,
    const float* __restrict__ b0, const float* __restrict__ b1,
    const float* __restrict__ b2, _Float16* __restrict__ q16,
    _Float16* __restrict__ k16, _Float16* __restrict__ v16) {
  __shared__ _Float16 Xs[2][128 * 32];
  __shared__ _Float16 Ws[2][128 * 32];

  int z = blockIdx.z;
  const float* W = (z == 0) ? W0 : (z == 1) ? W1 : W2;
  const float* bias = (z == 0) ? b0 : (z == 1) ? b1 : b2;
  _Float16* dst = (z == 0) ? q16 : (z == 1) ? k16 : v16;

  int m0 = blockIdx.x * 128;
  int n0 = blockIdx.y * 128;
  int tid = threadIdx.x;
  int wid = tid >> 5, lane = tid & 31;
  int lm = lane & 15, lh = lane >> 4;
  int wm = (wid & 3) * 32;   // wave M offset
  int wn = (wid >> 2) * 64;  // wave N offset

  float4 xr[4], wr[4];  // staged global data for next chunk

  auto load_global = [&](int k0) {
    for (int it = 0; it < 4; it++) {  // all loads issued before any use
      int v = tid + it * 256;
      int row = v >> 3, c4 = (v & 7) << 2;
      xr[it] = *(const float4*)(X + (size_t)(m0 + row) * DMODEL + k0 + c4);
      int kk = v >> 5, n4 = (v & 31) << 2;
      wr[it] = *(const float4*)(W + (size_t)(k0 + kk) * DMODEL + n0 + n4);
    }
  };
  auto store_lds = [&](int buf) {
    for (int it = 0; it < 4; it++) {
      int v = tid + it * 256;
      int row = v >> 3, c4 = (v & 7) << 2;
      union { _Float16 h[4]; v2u u; } pk;
      pk.h[0] = (_Float16)xr[it].x; pk.h[1] = (_Float16)xr[it].y;
      pk.h[2] = (_Float16)xr[it].z; pk.h[3] = (_Float16)xr[it].w;
      *(v2u*)(&Xs[buf][row * 32 + c4]) = pk.u;
      int kk = v >> 5, n4 = (v & 31) << 2;
      Ws[buf][(n4 + 0) * 32 + kk] = (_Float16)wr[it].x;
      Ws[buf][(n4 + 1) * 32 + kk] = (_Float16)wr[it].y;
      Ws[buf][(n4 + 2) * 32 + kk] = (_Float16)wr[it].z;
      Ws[buf][(n4 + 3) * 32 + kk] = (_Float16)wr[it].w;
    }
  };

  load_global(0);
  store_lds(0);
  v8f acc[2][4] = {};

  const int NKC = DMODEL / 32;  // 24
  for (int kc = 0; kc < NKC; kc++) {
    __syncthreads();
    if (kc + 1 < NKC) load_global((kc + 1) * 32);  // overlap with WMMAs
    int cb = kc & 1;
    Frag af[2], bf[4];
    for (int mt = 0; mt < 2; mt++) {
      const _Float16* p = &Xs[cb][(wm + mt * 16 + lm) * 32 + lh * 8];
      af[mt].u[0] = *(const v4u*)p;
      af[mt].u[1] = *(const v4u*)(p + 16);
    }
    for (int nt = 0; nt < 4; nt++) {
      const _Float16* p = &Ws[cb][(wn + nt * 16 + lm) * 32 + lh * 8];
      bf[nt].u[0] = *(const v4u*)p;
      bf[nt].u[1] = *(const v4u*)(p + 16);
    }
    for (int mt = 0; mt < 2; mt++)
      for (int nt = 0; nt < 4; nt++)
        acc[mt][nt] = wmma_f16(af[mt].h, bf[nt].h, acc[mt][nt]);
    if (kc + 1 < NKC) store_lds((kc + 1) & 1);
  }

  // epilogue: +bias, scatter to (b*H+h, s, e) f16
  for (int mt = 0; mt < 2; mt++)
    for (int nt = 0; nt < 4; nt++) {
      int col = n0 + wn + nt * 16 + lm;
      float bvv = bias[col];
      int h = col >> 6, e = col & 63;
      for (int r = 0; r < 8; r++) {
        int t = m0 + wm + mt * 16 + r + 8 * lh;  // token index = s*B + b
        int s = t >> 3, b = t & 7;
        dst[((size_t)(b * NHEAD + h) * S_LEN + s) * HD + e] =
            (_Float16)(acc[mt][nt][r] + bvv);
      }
    }
}

// ---------------------------------------------------------------------------
// Kernel 2: local windowed attention with look-around, flash-style online
// softmax, all matmuls via v_wmma_f32_16x16x32_f16. K tile staged with
// GLOBAL_LOAD_ASYNC_TO_LDS_B128 when the toolchain exposes it.
// grid (NWIN, B*H), 256 threads (8 waves x 16 query rows).
// ---------------------------------------------------------------------------
__global__ __launch_bounds__(256) void local_attn_wmma(
    const _Float16* __restrict__ q16, const _Float16* __restrict__ k16,
    const _Float16* __restrict__ v16, const int* __restrict__ kpm,
    float* __restrict__ out) {
  extern __shared__ char smem[];
  _Float16* Ksm = (_Float16*)smem;         // 384 x 64
  _Float16* Vts = Ksm + W3 * HD;           // 64 x 384 (V transposed)
  float* mJ = (float*)(Vts + HD * W3);     // 384
  float* mI = mJ + W3;                     // 128
  _Float16* Pst = (_Float16*)(mI + WIN);   // 8 waves * 16*32

  int w = blockIdx.x, bh = blockIdx.y;
  int b = bh / NHEAD, hh = bh % NHEAD;
  int tid = threadIdx.x, wid = tid >> 5, lane = tid & 31;
  int lm = lane & 15, lh = lane >> 4;
  const _Float16* Kg = k16 + (size_t)bh * S_LEN * HD;
  const _Float16* Vg = v16 + (size_t)bh * S_LEN * HD;

  // K tile (zero-padded look-around), row-major; async copy where possible
  for (int idx = tid; idx < W3 * 8; idx += 256) {
    int j = idx >> 3, c = idx & 7;
    int s = (w - 1) * WIN + j;
    _Float16* dp = Ksm + j * HD + c * 8;
    if (s >= 0 && s < S_LEN) {
#ifdef HAVE_ASYNC_LDS
      __builtin_amdgcn_global_load_async_to_lds_b128(
          (gv4i_p)(Kg + (size_t)s * HD + c * 8), (lv4i_p)dp, 0, 0);
#else
      *(v4u*)dp = *(const v4u*)(Kg + (size_t)s * HD + c * 8);
#endif
    } else {
      v4u zz = {};
      *(v4u*)dp = zz;
    }
  }
  for (int j = tid; j < W3; j += 256) {
    int s = (w - 1) * WIN + j;
    mJ[j] = (s < 0 || s >= S_LEN) ? 1.f : (kpm[b * S_LEN + s] ? 1.f : 0.f);
  }
  for (int i = tid; i < WIN; i += 256)
    mI[i] = kpm[b * S_LEN + w * WIN + i] ? 1.f : 0.f;
  // V tile transposed: Vts[e][j] (vector global read, scalar LDS scatter)
  for (int idx = tid; idx < W3 * 8; idx += 256) {
    int j = idx >> 3, c = idx & 7;
    int s = (w - 1) * WIN + j;
    _Float16 vh[8];
    if (s >= 0 && s < S_LEN) {
      *(v4u*)vh = *(const v4u*)(Vg + (size_t)s * HD + c * 8);
    } else {
      v4u zz = {};
      *(v4u*)vh = zz;
    }
    for (int q = 0; q < 8; q++) Vts[(c * 8 + q) * W3 + j] = vh[q];
  }
  wait_async0();
  __syncthreads();

  int i0 = wid * 16;
  const _Float16* Qg = q16 + ((size_t)bh * S_LEN + (size_t)w * WIN + i0) * HD;
  Frag aq[2];
  for (int ec = 0; ec < 2; ec++) {
    const _Float16* p = Qg + lm * HD + ec * 32 + lh * 8;
    aq[ec].u[0] = *(const v4u*)p;
    aq[ec].u[1] = *(const v4u*)(p + 16);
  }

  const float inv = 0.125f;  // hd^-0.5
  v8f Oacc[4] = {};
  float mr[8], lr[8], mi_row[8];
  for (int r = 0; r < 8; r++) {
    mr[r] = NEGMAX;
    lr[r] = 0.f;
    mi_row[r] = mI[i0 + r + 8 * lh];
  }
  _Float16* Pw = Pst + wid * 512;

  for (int jt = 0; jt < 12; jt++) {
    // scores: two 16-col tiles over hd (K = 2 chunks of 32)
    v8f c0 = {}, c1 = {};
    for (int ec = 0; ec < 2; ec++) {
      Frag bk0, bk1;
      const _Float16* p0 = Ksm + (jt * 32 + lm) * HD + ec * 32 + lh * 8;
      bk0.u[0] = *(const v4u*)p0; bk0.u[1] = *(const v4u*)(p0 + 16);
      const _Float16* p1 = Ksm + (jt * 32 + 16 + lm) * HD + ec * 32 + lh * 8;
      bk1.u[0] = *(const v4u*)p1; bk1.u[1] = *(const v4u*)(p1 + 16);
      c0 = wmma_f16(aq[ec].h, bk0.h, c0);
      c1 = wmma_f16(aq[ec].h, bk1.h, c1);
    }
    float mj0 = mJ[jt * 32 + lm];
    float mj1 = mJ[jt * 32 + 16 + lm];
    float scl[8];
    for (int r = 0; r < 8; r++) {
      float s0 = (mi_row[r] + mj0 > 0.f) ? NEGMAX : c0[r] * inv;
      float s1 = (mi_row[r] + mj1 > 0.f) ? NEGMAX : c1[r] * inv;
      float tm = fmaxf(s0, s1);
      for (int off = 1; off < 16; off <<= 1)
        tm = fmaxf(tm, __shfl_xor(tm, off, 32));
      float mn = fmaxf(mr[r], tm);
      float sc = __expf(mr[r] - mn);
      mr[r] = mn;
      float p0 = __expf(s0 - mn);
      float p1 = __expf(s1 - mn);
      float ps = p0 + p1;
      for (int off = 1; off < 16; off <<= 1) ps += __shfl_xor(ps, off, 32);
      lr[r] = lr[r] * sc + ps;
      scl[r] = sc;
      c0[r] = p0;
      c1[r] = p1;
    }
    for (int t = 0; t < 4; t++)
      for (int r = 0; r < 8; r++) Oacc[t][r] *= scl[r];

    // stage P (16x32 f16) row-major through per-wave LDS, reload as A-frag
    for (int r = 0; r < 8; r++) {
      int il = r + 8 * lh;
      Pw[il * 32 + lm] = (_Float16)c0[r];
      Pw[il * 32 + 16 + lm] = (_Float16)c1[r];
    }
    asm volatile("" ::: "memory");  // per-wave DS ops are HW in-order
    Frag ap;
    const _Float16* pp = Pw + lm * 32 + lh * 8;
    ap.u[0] = *(const v4u*)pp;
    ap.u[1] = *(const v4u*)(pp + 16);
    for (int t = 0; t < 4; t++) {
      Frag bvf;
      const _Float16* vp = Vts + (t * 16 + lm) * W3 + jt * 32 + lh * 8;
      bvf.u[0] = *(const v4u*)vp;
      bvf.u[1] = *(const v4u*)(vp + 16);
      Oacc[t] = wmma_f16(ap.h, bvf.h, Oacc[t]);
    }
  }

  for (int t = 0; t < 4; t++) {
    int col = hh * HD + t * 16 + lm;
    for (int r = 0; r < 8; r++) {
      int srow = w * WIN + i0 + r + 8 * lh;
      out[((size_t)srow * BATCH + b) * DMODEL + col] = Oacc[t][r] / lr[r];
    }
  }
}

// ---------------------------------------------------------------------------
// Kernel 3: gq = hidden[0] @ Wgq + bgq  (tiny)
// ---------------------------------------------------------------------------
__global__ __launch_bounds__(256) void gq_proj(const float* __restrict__ X,
                                               const float* __restrict__ Wgq,
                                               const float* __restrict__ bgq,
                                               float* __restrict__ gq) {
  int idx = blockIdx.x * 256 + threadIdx.x;
  if (idx >= BATCH * DMODEL) return;
  int b = idx / DMODEL, d = idx % DMODEL;
  const float* xr = X + (size_t)b * DMODEL;  // hidden[(0,b),:]
  float s = bgq[d];
  for (int k = 0; k < DMODEL; k++) s += xr[k] * Wgq[(size_t)k * DMODEL + d];
  gq[idx] = s;  // layout (b*H+h)*64+e == b*768+d
}

// ---------------------------------------------------------------------------
// Kernel 4: global-token attention, overwrites out row 0. Deterministic
// tree reductions (no float atomics).
// ---------------------------------------------------------------------------
__global__ __launch_bounds__(256) void global_attn_kernel(
    const float* __restrict__ gq, const _Float16* __restrict__ k16,
    const _Float16* __restrict__ v16, const int* __restrict__ kpm,
    float* __restrict__ out) {
  __shared__ float sc[S_LEN];
  __shared__ float qsh[HD];
  __shared__ float red[256];
  __shared__ float accv[HD + 1];
  int bh = blockIdx.x;
  int b = bh / NHEAD, hh = bh % NHEAD;
  int tid = threadIdx.x;
  if (tid < HD) qsh[tid] = gq[bh * HD + tid];
  __syncthreads();

  const _Float16* Kg = k16 + (size_t)bh * S_LEN * HD;
  const _Float16* Vg = v16 + (size_t)bh * S_LEN * HD;
  float lmax = NEGMAX;
  for (int s = tid; s < S_LEN; s += 256) {
    const _Float16* kr = Kg + (size_t)s * HD;
    float d = 0.f;
    for (int e = 0; e < HD; e++) d += qsh[e] * (float)kr[e];
    d *= 0.125f;
    if (kpm[b * S_LEN + s]) d = NEGMAX;
    sc[s] = d;
    lmax = fmaxf(lmax, d);
  }
  red[tid] = lmax;
  __syncthreads();
  for (int st = 128; st > 0; st >>= 1) {
    if (tid < st) red[tid] = fmaxf(red[tid], red[tid + st]);
    __syncthreads();
  }
  float M = red[0];
  __syncthreads();

  float av[HD];
  for (int e = 0; e < HD; e++) av[e] = 0.f;
  float lsum = 0.f;
  for (int s = tid; s < S_LEN; s += 256) {
    float p = __expf(sc[s] - M);
    lsum += p;
    const _Float16* vr = Vg + (size_t)s * HD;
    for (int e = 0; e < HD; e++) av[e] += p * (float)vr[e];
  }
  for (int e = 0; e < HD; e++) {
    red[tid] = av[e];
    __syncthreads();
    for (int st = 128; st > 0; st >>= 1) {
      if (tid < st) red[tid] += red[tid + st];
      __syncthreads();
    }
    if (tid == 0) accv[e] = red[0];
    __syncthreads();
  }
  red[tid] = lsum;
  __syncthreads();
  for (int st = 128; st > 0; st >>= 1) {
    if (tid < st) red[tid] += red[tid + st];
    __syncthreads();
  }
  if (tid == 0) accv[HD] = red[0];
  __syncthreads();
  if (tid < HD)
    out[(size_t)b * DMODEL + hh * HD + tid] = accv[tid] / accv[HD];
}

// ---------------------------------------------------------------------------
extern "C" void kernel_launch(void* const* d_in, const int* in_sizes, int n_in,
                              void* d_out, int out_size, void* d_ws,
                              size_t ws_size, hipStream_t stream) {
  (void)in_sizes; (void)n_in; (void)out_size; (void)ws_size;
  const float* hs  = (const float*)d_in[0];
  const int*   kpm = (const int*)d_in[1];
  const float* Wq  = (const float*)d_in[2];
  const float* Wk  = (const float*)d_in[3];
  const float* Wv  = (const float*)d_in[4];
  const float* Wgq = (const float*)d_in[5];
  const float* bq  = (const float*)d_in[6];
  const float* bk  = (const float*)d_in[7];
  const float* bv  = (const float*)d_in[8];
  const float* bgq = (const float*)d_in[9];
  float* out = (float*)d_out;

  size_t per = (size_t)BATCH * NHEAD * S_LEN * HD;  // halfs per q/k/v tensor
  _Float16* q16 = (_Float16*)d_ws;
  _Float16* k16 = q16 + per;
  _Float16* v16 = k16 + per;
  float* gqbuf = (float*)(v16 + per);

  dim3 g1(256, 6, 3);
  qkv_gemm_wmma<<<g1, 256, 0, stream>>>(hs, Wq, Wk, Wv, bq, bk, bv, q16, k16,
                                        v16);
  gq_proj<<<(BATCH * DMODEL + 255) / 256, 256, 0, stream>>>(hs, Wgq, bgq,
                                                            gqbuf);
  size_t lds_bytes = (size_t)W3 * HD * 2 + (size_t)HD * W3 * 2 + W3 * 4 +
                     WIN * 4 + (size_t)8 * 512 * 2;  // 108,544 B (<320KB WGP)
  dim3 g2(NWIN, BATCH * NHEAD);
  local_attn_wmma<<<g2, 256, lds_bytes, stream>>>(q16, k16, v16, kpm, out);
  global_attn_kernel<<<BATCH * NHEAD, 256, 0, stream>>>(gqbuf, k16, v16, kpm,
                                                        out);
}
